// GlobalSubSampleAttn_20040317403634
// MI455X (gfx1250) — compile-verified, matrix operand order checked by reading, not run on hardware
//
#include <hip/hip_runtime.h>
#include <stdint.h>
#include <stddef.h>

// ---------------- problem dims (hardcoded from reference) ----------------
static constexpr int Bsz  = 4;
static constexpr int Hh   = 128;
static constexpr int Wh   = 128;
static constexpr int Cc   = 256;
static constexpr int Nn   = Hh * Wh;          // 16384
static constexpr int Ssz  = (Hh/2) * (Wh/2);  // 4096
static constexpr int NH   = 8;
static constexpr int ROWS_X = Bsz * Nn;       // 65536
static constexpr int ROWS_S = Bsz * Ssz;      // 16384

// ---------------- types ----------------
typedef __bf16 bf16_t;
typedef __attribute__((ext_vector_type(16))) __bf16 v16bf;
typedef __attribute__((ext_vector_type(8)))  __bf16 v8bf;
typedef __attribute__((ext_vector_type(8)))  float  v8f;

struct bf4 { bf16_t e[4]; };   // POD 8-byte pack

__device__ inline bf16_t f2bf(float f) {
  union { float f; unsigned u; } a; a.f = f;
  unsigned r = a.u + 0x7FFFu + ((a.u >> 16) & 1u);
  unsigned short h = (unsigned short)(r >> 16);
  union { unsigned short u; bf16_t b; } o; o.u = h;
  return o.b;
}
__device__ inline float bf2f(bf16_t b) {
  union { unsigned short u; bf16_t b; } i; i.b = b;
  union { unsigned u; float f; } o; o.u = ((unsigned)i.u) << 16;
  return o.f;
}

// Assemble 16-elem bf16 WMMA fragment: 8 contiguous at p, 8 at p+16.
__device__ inline v16bf frag16(const bf16_t* p) {
  v8bf lo = *(const v8bf*)p;
  v8bf hi = *(const v8bf*)(p + 16);
  return __builtin_shufflevector(lo, hi, 0,1,2,3,4,5,6,7,8,9,10,11,12,13,14,15);
}

// gfx1250 async global->LDS copy, 16B per lane, tracked by ASYNCcnt.
// LDS dest VGPR = wave-relative LDS byte offset (low 32b of flat LDS address).
__device__ inline uint32_t lds_off32(const void* p) {
  return (uint32_t)(uintptr_t)p;
}
#define ASYNC_LD_B128(ldsoff, gptr)                                        \
  asm volatile("global_load_async_to_lds_b128 %0, %1, off"                 \
               :: "v"(ldsoff), "v"((uint64_t)(uintptr_t)(gptr)) : "memory")

// ---------------- weight prep ----------------
__global__ void k_transpose_w(const float* __restrict__ w, bf16_t* __restrict__ wT,
                              int K, int Ncol) {
  int idx = blockIdx.x * blockDim.x + threadIdx.x;
  if (idx >= K * Ncol) return;
  int k = idx / Ncol, n = idx % Ncol;
  wT[(size_t)n * K + k] = f2bf(w[idx]);
}

__global__ void k_pack_srw(const float* __restrict__ w, bf16_t* __restrict__ wT) {
  int idx = blockIdx.x * blockDim.x + threadIdx.x;
  if (idx >= Cc * 4 * Cc) return;
  int o = idx >> 10, r = idx & 1023;
  int p = r >> 8, i = r & 255;
  int kh = p >> 1, kw = p & 1;
  wT[(size_t)o * 1024 + r] = f2bf(w[(((size_t)o * Cc + i) * 2 + kh) * 2 + kw]);
}

__global__ void k_pack_x(const float* __restrict__ x, bf16_t* __restrict__ cat) {
  int idx = blockIdx.x * blockDim.x + threadIdx.x;   // one per 4 elems
  long e = (long)idx * 4;
  if (e >= (long)ROWS_X * Cc) return;
  int row = (int)(e >> 8), c = (int)(e & 255);
  float4 v = *(const float4*)(x + (size_t)row * Cc + c);
  bf4 pk; pk.e[0]=f2bf(v.x); pk.e[1]=f2bf(v.y); pk.e[2]=f2bf(v.z); pk.e[3]=f2bf(v.w);
  *(bf4*)(cat + (size_t)row * 512 + c) = pk;
}

__global__ void k_im2col(const float* __restrict__ x, bf16_t* __restrict__ xim) {
  int bs = blockIdx.x;                  // 0..ROWS_S-1
  int t  = threadIdx.x;                 // 256 threads * 4 elems = 1024
  int b  = bs >> 12;
  int s  = bs & 4095;
  int oh = s >> 6, ow = s & 63;
  int kidx = t * 4;
  int p = kidx >> 8, i = kidx & 255;
  int kh = p >> 1, kw = p & 1;
  int n = (2*oh + kh) * Wh + (2*ow + kw);
  float4 v = *(const float4*)(x + ((size_t)b * Nn + n) * Cc + i);
  bf4 pk; pk.e[0]=f2bf(v.x); pk.e[1]=f2bf(v.y); pk.e[2]=f2bf(v.z); pk.e[3]=f2bf(v.w);
  *(bf4*)(xim + (size_t)bs * 1024 + kidx) = pk;
}

// ---------------- generic WMMA GEMM (128x64 tile, async double-buffered) ----------------
enum { EP_F32 = 0, EP_BIAS_F32 = 1, EP_ELU_BF = 2, EP_SCALE_BF = 3, EP_RELU_BF = 4 };

// C[M,Ncol] = epi( A[M,K](lda) * BT[Ncol,K]^T ), block tile 128x64, k-step 32.
// 8 waves: wave -> (wy=w>>1 in 0..3 : 32 rows, wx=w&1 : 32 cols), 2x2 accs each.
__global__ __launch_bounds__(256) void k_gemm_wmma(
    const bf16_t* __restrict__ A, int lda,
    const bf16_t* __restrict__ BT,
    int M, int Ncol, int K,
    float* __restrict__ outF, bf16_t* __restrict__ outB, int ldo,
    const float* __restrict__ bias, float scale, int mode)
{
  __shared__ __align__(16) bf16_t As[2][128 * 32];
  __shared__ __align__(16) bf16_t Bs[2][64 * 32];
  const int tid  = threadIdx.x;
  const int lane = tid & 31, wave = tid >> 5;
  const int wy = wave >> 1, wx = wave & 1;
  const int row0 = blockIdx.y * 128;
  const int col0 = blockIdx.x * 64;
  // tile loaders: A 128x32 -> 16 bf16/thread (2x b128); B 64x32 -> 8 bf16/thread
  const int lrA = tid >> 1, lkA = (tid & 1) << 4;
  const int lrB = tid >> 2, lkB = (tid & 3) << 3;
  const bf16_t* Ag = A  + (size_t)(row0 + lrA) * lda + lkA;
  const bf16_t* Bg = BT + (size_t)(col0 + lrB) * K   + lkB;
  uint32_t ldsA0[2], ldsA1[2], ldsBo[2];
#pragma unroll
  for (int b = 0; b < 2; ++b) {
    ldsA0[b] = lds_off32(&As[b][lrA * 32 + lkA]);
    ldsA1[b] = lds_off32(&As[b][lrA * 32 + lkA + 8]);
    ldsBo[b] = lds_off32(&Bs[b][lrB * 32 + lkB]);
  }
  const int r = lane & 15, hb = (lane >> 4) << 3;

  v8f acc[2][2] = {};

  // prologue: stage k0=0 into buffer 0
  ASYNC_LD_B128(ldsA0[0], Ag);
  ASYNC_LD_B128(ldsA1[0], Ag + 8);
  ASYNC_LD_B128(ldsBo[0], Bg);

  for (int k0 = 0; k0 < K; k0 += 32) {
    const int cur = (k0 >> 5) & 1;
    if (k0 + 32 < K) {              // stage next tile into other buffer
      ASYNC_LD_B128(ldsA0[cur ^ 1], Ag + k0 + 32);
      ASYNC_LD_B128(ldsA1[cur ^ 1], Ag + k0 + 40);
      ASYNC_LD_B128(ldsBo[cur ^ 1], Bg + k0 + 32);
      asm volatile("s_wait_asynccnt 0x3" ::: "memory");   // current 3 done
    } else {
      asm volatile("s_wait_asynccnt 0x0" ::: "memory");
    }
    __syncthreads();
    const bf16_t* as = &As[cur][0];
    const bf16_t* bs = &Bs[cur][0];
    v16bf a0 = frag16(as + (wy * 32 +      r) * 32 + hb);
    v16bf a1 = frag16(as + (wy * 32 + 16 + r) * 32 + hb);
    v16bf b0 = frag16(bs + (wx * 32 +      r) * 32 + hb);
    v16bf b1 = frag16(bs + (wx * 32 + 16 + r) * 32 + hb);
    acc[0][0] = __builtin_amdgcn_wmma_f32_16x16x32_bf16(false, a0, false, b0,
                                                        (short)0, acc[0][0], false, false);
    acc[0][1] = __builtin_amdgcn_wmma_f32_16x16x32_bf16(false, a0, false, b1,
                                                        (short)0, acc[0][1], false, false);
    acc[1][0] = __builtin_amdgcn_wmma_f32_16x16x32_bf16(false, a1, false, b0,
                                                        (short)0, acc[1][0], false, false);
    acc[1][1] = __builtin_amdgcn_wmma_f32_16x16x32_bf16(false, a1, false, b1,
                                                        (short)0, acc[1][1], false, false);
    __syncthreads();
  }

  const int half = lane >> 4;
#pragma unroll
  for (int m = 0; m < 2; ++m) {
#pragma unroll
    for (int i = 0; i < 8; ++i) {
      int orow = row0 + wy * 32 + m * 16 + i + half * 8;
#pragma unroll
      for (int n = 0; n < 2; ++n) {
        int ocol = col0 + wx * 32 + n * 16 + r;
        float v = acc[m][n][i];
        if (mode == EP_F32) {
          outF[(size_t)orow * ldo + ocol] = v;
        } else if (mode == EP_BIAS_F32) {
          outF[(size_t)orow * ldo + ocol] = v + bias[ocol];
        } else if (mode == EP_ELU_BF) {   // elu(x)+1 : x+1 if x>0 else exp(x)
          outB[(size_t)orow * ldo + ocol] = f2bf(v > 0.f ? v + 1.f : __expf(v));
        } else if (mode == EP_SCALE_BF) {
          outB[(size_t)orow * ldo + ocol] = f2bf(v * scale);
        } else {                          // relu -> bf16
          outB[(size_t)orow * ldo + ocol] = f2bf(v > 0.f ? v : 0.f);
        }
      }
    }
  }
}

// ---------------- LayerNorm over 256-wide rows ----------------
__global__ __launch_bounds__(256) void k_layernorm(
    const float* __restrict__ in, const float* __restrict__ g, const float* __restrict__ bb,
    bf16_t* __restrict__ outB, int ldoB,
    float* __restrict__ outF, const float* __restrict__ resid, int rows)
{
  int wave = threadIdx.x >> 5, lane = threadIdx.x & 31;
  int row = blockIdx.x * 8 + wave;
  if (row >= rows) return;
  const float* p = in + (size_t)row * 256;
  float4 a = *(const float4*)(p + lane * 8);
  float4 c = *(const float4*)(p + lane * 8 + 4);
  float vals[8] = {a.x, a.y, a.z, a.w, c.x, c.y, c.z, c.w};
  float s = 0.f;
#pragma unroll
  for (int j = 0; j < 8; ++j) s += vals[j];
#pragma unroll
  for (int o = 16; o > 0; o >>= 1) s += __shfl_xor(s, o, 32);
  float mean = s * (1.f / 256.f);
  float vs = 0.f;
#pragma unroll
  for (int j = 0; j < 8; ++j) { vals[j] -= mean; vs += vals[j] * vals[j]; }
#pragma unroll
  for (int o = 16; o > 0; o >>= 1) vs += __shfl_xor(vs, o, 32);
  float inv = rsqrtf(vs * (1.f / 256.f) + 1e-5f);
#pragma unroll
  for (int j = 0; j < 8; ++j) {
    int c0 = lane * 8 + j;
    float y = vals[j] * inv * g[c0] + bb[c0];
    if (outB) outB[(size_t)row * ldoB + c0] = f2bf(y);
    if (outF) outF[(size_t)row * 256 + c0] = y + resid[(size_t)row * 256 + c0];
  }
}

// ---------------- Ksum = sum_s K_elu ----------------
__global__ void k_ksum_part(const bf16_t* __restrict__ Ke, float* __restrict__ part) {
  int blk = blockIdx.x;                 // B*16
  int b = blk >> 4, ch = blk & 15;
  int c = threadIdx.x;                  // 256
  const bf16_t* p = Ke + ((size_t)b * Ssz + ch * 256) * 256 + c;
  float s = 0.f;
  for (int i = 0; i < 256; ++i) s += bf2f(p[(size_t)i * 256]);
  part[((size_t)b * 16 + ch) * 256 + c] = s;
}
__global__ void k_ksum_reduce(const float* __restrict__ part, float* __restrict__ ksum) {
  int idx = blockIdx.x * blockDim.x + threadIdx.x;  // B*256
  if (idx >= Bsz * 256) return;
  int b = idx >> 8, c = idx & 255;
  float s = 0.f;
  for (int ch = 0; ch < 16; ++ch) s += part[((size_t)b * 16 + ch) * 256 + c];
  ksum[idx] = s;
}

// ---------------- KV[b,h] = K^T * Vsc (32x32 over S=4096), stored transposed ----------------
__global__ __launch_bounds__(256) void k_kv_wmma(const bf16_t* __restrict__ Ke,
                                                 const bf16_t* __restrict__ Vsc,
                                                 bf16_t* __restrict__ KVbT)
{
  __shared__ __align__(16) bf16_t KT[32 * 32];  // [d][s]
  __shared__ __align__(16) bf16_t VT[32 * 32];  // [e][s]
  int bh = blockIdx.x;                  // B*NH
  int b = bh >> 3, h = bh & 7;
  int tid = threadIdx.x, lane = tid & 31, wave = tid >> 5;
  int d0 = (tid & 7) * 4;
  int si = tid >> 3;
  const int r = lane & 15, hb = (lane >> 4) << 3;
  int ti = (wave >> 1) & 1, tj = wave & 1;
  v8f acc = {0,0,0,0,0,0,0,0};
  const size_t base = (size_t)b * Ssz * 256 + h * 32;

  for (int s0 = 0; s0 < Ssz; s0 += 32) {
    bf4 kk = *(const bf4*)(Ke  + base + (size_t)(s0 + si) * 256 + d0);
    bf4 vv = *(const bf4*)(Vsc + base + (size_t)(s0 + si) * 256 + d0);
    __syncthreads();
#pragma unroll
    for (int j = 0; j < 4; ++j) {
      KT[(d0 + j) * 32 + si] = kk.e[j];
      VT[(d0 + j) * 32 + si] = vv.e[j];
    }
    __syncthreads();
    if (wave < 4) {
      v16bf af = frag16(KT + (ti * 16 + r) * 32 + hb);
      v16bf bf = frag16(VT + (tj * 16 + r) * 32 + hb);
      acc = __builtin_amdgcn_wmma_f32_16x16x32_bf16(false, af, false, bf,
                                                    (short)0, acc, false, false);
    }
  }
  if (wave < 4) {
    int half = lane >> 4;
#pragma unroll
    for (int i = 0; i < 8; ++i) {
      int d = ti * 16 + i + half * 8;
      int e = tj * 16 + r;
      KVbT[((size_t)bh * 32 + e) * 32 + d] = f2bf(acc[i]);
    }
  }
}

// ---------------- msg = (Q * KV) * S/(Q.Ksum + eps) ----------------
__global__ __launch_bounds__(256) void k_msg_wmma(const bf16_t* __restrict__ Qe,
                                                  const bf16_t* __restrict__ KVbT,
                                                  const float* __restrict__ ksum,
                                                  bf16_t* __restrict__ msgb)
{
  int row0 = blockIdx.x * 16;
  int lane = threadIdx.x & 31, h = threadIdx.x >> 5;
  int b = row0 >> 14;
  const int r = lane & 15, hb = (lane >> 4) << 3;

  const bf16_t* Ap = Qe + (size_t)(row0 + r) * 256 + h * 32;
  v16bf af = frag16(Ap + hb);
  const bf16_t* Bp = KVbT + (size_t)(b * 8 + h) * 32 * 32;
  v16bf bf0 = frag16(Bp + r * 32 + hb);
  v16bf bf1 = frag16(Bp + (16 + r) * 32 + hb);
  v8f acc0 = {0,0,0,0,0,0,0,0};
  v8f acc1 = {0,0,0,0,0,0,0,0};
  acc0 = __builtin_amdgcn_wmma_f32_16x16x32_bf16(false, af, false, bf0,
                                                 (short)0, acc0, false, false);
  acc1 = __builtin_amdgcn_wmma_f32_16x16x32_bf16(false, af, false, bf1,
                                                 (short)0, acc1, false, false);

  const float* ks = ksum + (size_t)b * 256 + h * 32;
  float dot = 0.f;
#pragma unroll
  for (int d = 0; d < 32; ++d) dot += bf2f(Ap[d]) * ks[d];
  float z = (float)Ssz / (dot + 1e-6f);

  int half = lane >> 4;
#pragma unroll
  for (int i = 0; i < 8; ++i) {
    float zi = __shfl(z, i + half * 8, 32);
    int orow = row0 + i + half * 8;
    bf16_t* op = msgb + (size_t)orow * 256 + h * 32;
    op[r]      = f2bf(acc0[i] * zi);
    op[16 + r] = f2bf(acc1[i] * zi);
  }
}

// ---------------- host launcher ----------------
extern "C" void kernel_launch(void* const* d_in, const int* in_sizes, int n_in,
                              void* d_out, int out_size, void* d_ws, size_t ws_size,
                              hipStream_t stream) {
  const float* x      = (const float*)d_in[0];
  const float* sr_w   = (const float*)d_in[1];
  const float* sr_b   = (const float*)d_in[2];
  const float* norm_g = (const float*)d_in[3];
  const float* norm_b = (const float*)d_in[4];
  const float* wq     = (const float*)d_in[5];
  const float* wk     = (const float*)d_in[6];
  const float* wv     = (const float*)d_in[7];
  const float* wm     = (const float*)d_in[8];
  const float* w1     = (const float*)d_in[9];
  const float* w2     = (const float*)d_in[10];
  const float* n1g    = (const float*)d_in[11];
  const float* n1b    = (const float*)d_in[12];
  const float* n2g    = (const float*)d_in[13];
  const float* n2b    = (const float*)d_in[14];
  float* out = (float*)d_out;

  char* ws = (char*)d_ws;
  size_t off = 0;
  auto alloc = [&](size_t bytes) -> void* {
    off = (off + 255) & ~(size_t)255;
    void* p = ws + off;
    off += bytes;
    return p;
  };

  bf16_t* wqT  = (bf16_t*)alloc((size_t)256 * 256 * 2);
  bf16_t* wkT  = (bf16_t*)alloc((size_t)256 * 256 * 2);
  bf16_t* wvT  = (bf16_t*)alloc((size_t)256 * 256 * 2);
  bf16_t* wmT  = (bf16_t*)alloc((size_t)256 * 256 * 2);
  bf16_t* w1T  = (bf16_t*)alloc((size_t)512 * 512 * 2);
  bf16_t* w2T  = (bf16_t*)alloc((size_t)256 * 512 * 2);
  bf16_t* wsrT = (bf16_t*)alloc((size_t)256 * 1024 * 2);
  float*  part = (float*)alloc((size_t)Bsz * 16 * 256 * 4);
  float*  ksum = (float*)alloc((size_t)Bsz * 256 * 4);
  bf16_t* KVbT = (bf16_t*)alloc((size_t)Bsz * NH * 32 * 32 * 2);
  bf16_t* cat  = (bf16_t*)alloc((size_t)ROWS_X * 512 * 2);          // [x | msg_ln]
  // pool A: im2col buf, later reused for msg (both 32 MiB)
  bf16_t* xim  = (bf16_t*)alloc((size_t)ROWS_S * 1024 * 2);
  bf16_t* msgb = xim;
  // pool B: Qe + xs + xs_ln + Ke + Vsc, later reused for h1 (64 MiB <= 72 MiB)
  size_t szQe = (size_t)ROWS_X * 256 * 2;
  size_t szXs = (size_t)ROWS_S * 256 * 4;
  size_t szLn = (size_t)ROWS_S * 256 * 2;
  char* poolB = (char*)alloc(szQe + szXs + 3 * szLn);
  bf16_t* Qe    = (bf16_t*)poolB;
  float*  xs    = (float*)(poolB + szQe);
  bf16_t* xs_ln = (bf16_t*)(poolB + szQe + szXs);
  bf16_t* Ke    = (bf16_t*)(poolB + szQe + szXs + szLn);
  bf16_t* Vsc   = (bf16_t*)(poolB + szQe + szXs + 2 * szLn);
  bf16_t* h1    = (bf16_t*)poolB;        // [ROWS_X][512] bf16
  // pool C: mm f32, reused for h2 f32
  float* mm = (float*)alloc((size_t)ROWS_X * 256 * 4);
  float* h2 = mm;

  // --- weight prep ---
  k_transpose_w<<<(256 * 256 + 255) / 256, 256, 0, stream>>>(wq, wqT, 256, 256);
  k_transpose_w<<<(256 * 256 + 255) / 256, 256, 0, stream>>>(wk, wkT, 256, 256);
  k_transpose_w<<<(256 * 256 + 255) / 256, 256, 0, stream>>>(wv, wvT, 256, 256);
  k_transpose_w<<<(256 * 256 + 255) / 256, 256, 0, stream>>>(wm, wmT, 256, 256);
  k_transpose_w<<<(512 * 512 + 255) / 256, 256, 0, stream>>>(w1, w1T, 512, 512);
  k_transpose_w<<<(512 * 256 + 255) / 256, 256, 0, stream>>>(w2, w2T, 512, 256);
  k_pack_srw<<<(256 * 1024 + 255) / 256, 256, 0, stream>>>(sr_w, wsrT);
  k_pack_x<<<(ROWS_X * 256 / 4 + 255) / 256, 256, 0, stream>>>(x, cat);
  k_im2col<<<ROWS_S, 256, 0, stream>>>(x, xim);

  // --- spatial reduction conv (im2col GEMM) + LN ---
  k_gemm_wmma<<<dim3(256 / 64, ROWS_S / 128), 256, 0, stream>>>(
      xim, 1024, wsrT, ROWS_S, 256, 1024, xs, nullptr, 256, sr_b, 0.f, EP_BIAS_F32);
  k_layernorm<<<ROWS_S / 8, 256, 0, stream>>>(xs, norm_g, norm_b, xs_ln, 256,
                                              nullptr, nullptr, ROWS_S);

  // --- projections (Q from cat[:, :256], K/V from xs_ln) ---
  k_gemm_wmma<<<dim3(4, ROWS_X / 128), 256, 0, stream>>>(
      cat, 512, wqT, ROWS_X, 256, 256, nullptr, Qe, 256, nullptr, 0.f, EP_ELU_BF);
  k_gemm_wmma<<<dim3(4, ROWS_S / 128), 256, 0, stream>>>(
      xs_ln, 256, wkT, ROWS_S, 256, 256, nullptr, Ke, 256, nullptr, 0.f, EP_ELU_BF);
  k_gemm_wmma<<<dim3(4, ROWS_S / 128), 256, 0, stream>>>(
      xs_ln, 256, wvT, ROWS_S, 256, 256, nullptr, Vsc, 256, nullptr,
      1.f / (float)Ssz, EP_SCALE_BF);

  // --- linear attention ---
  k_ksum_part<<<Bsz * 16, 256, 0, stream>>>(Ke, part);
  k_ksum_reduce<<<(Bsz * 256 + 255) / 256, 256, 0, stream>>>(part, ksum);
  k_kv_wmma<<<Bsz * NH, 256, 0, stream>>>(Ke, Vsc, KVbT);
  k_msg_wmma<<<ROWS_X / 16, 256, 0, stream>>>(Qe, KVbT, ksum, msgb);

  // --- merge proj + LN1 (into cat[:, 256:512]) ---
  k_gemm_wmma<<<dim3(4, ROWS_X / 128), 256, 0, stream>>>(
      msgb, 256, wmT, ROWS_X, 256, 256, mm, nullptr, 256, nullptr, 0.f, EP_F32);
  k_layernorm<<<ROWS_X / 8, 256, 0, stream>>>(mm, n1g, n1b, cat + 256, 512,
                                              nullptr, nullptr, ROWS_X);

  // --- MLP ---
  k_gemm_wmma<<<dim3(8, ROWS_X / 128), 256, 0, stream>>>(
      cat, 512, w1T, ROWS_X, 512, 512, nullptr, h1, 512, nullptr, 0.f, EP_RELU_BF);
  k_gemm_wmma<<<dim3(4, ROWS_X / 128), 256, 0, stream>>>(
      h1, 512, w2T, ROWS_X, 256, 512, h2, nullptr, 256, nullptr, 0.f, EP_F32);

  // --- LN2 + residual ---
  k_layernorm<<<ROWS_X / 8, 256, 0, stream>>>(h2, n2g, n2b, nullptr, 0,
                                              out, x, ROWS_X);
}
